// FullAttention_755914244459
// MI455X (gfx1250) — compile-verified, matrix operand order checked by reading, not run on hardware
//
#include <hip/hip_runtime.h>

// ---------------------------------------------------------------------------
// Types
// ---------------------------------------------------------------------------
typedef unsigned short u16x4  __attribute__((ext_vector_type(4)));
typedef unsigned short u16x8  __attribute__((ext_vector_type(8)));
typedef unsigned short u16x16 __attribute__((ext_vector_type(16)));
typedef __bf16         v16bf  __attribute__((ext_vector_type(16)));
typedef float          v8f    __attribute__((ext_vector_type(8)));

#define S_LEN 2048
#define HID   1024
#define NHEAD 16
#define HDIM  64

// round-to-nearest-even f32 -> bf16 (bit pattern in ushort)
__device__ __forceinline__ unsigned short f2bf(float f) {
    unsigned int u = __float_as_uint(f);
    u += 0x7FFFu + ((u >> 16) & 1u);
    return (unsigned short)(u >> 16);
}

// WMMA bf16 wrapper: D = A(16x32) * B(32x16) + C, f32 accum
__device__ __forceinline__ v8f wmma_bf16(v16bf a, v16bf b, v8f c) {
    return __builtin_amdgcn_wmma_f32_16x16x32_bf16(
        /*neg_a=*/false, a, /*neg_b=*/false, b,
        /*c_mod=*/(short)0, c, /*reuse_a=*/false, /*reuse_b=*/false);
}

// ---------------------------------------------------------------------------
// CDNA5 async global -> LDS (fire-and-forget, tracked by ASYNCcnt).
// VGLOBAL GV form: vdst = LDS byte address (flat shared ptr low 32 bits per
// the ISA aperture rule); vaddr = 64-bit global address.
// ---------------------------------------------------------------------------
__device__ __forceinline__ void async_load_b128(const void* gptr, void* lptr) {
    unsigned lds_addr = (unsigned)(size_t)lptr;
    asm volatile("global_load_async_to_lds_b128 %0, %1, off"
                 :: "v"(lds_addr), "v"(gptr)
                 : "memory");
}
__device__ __forceinline__ void wait_asynccnt0() {
    asm volatile("s_wait_asynccnt 0" ::: "memory");
}

// Load a 16x32 bf16 fragment (A-role, or B-role as B^T) for this lane.
// rowp points at this lane's row (m = lane%16 already resolved by caller),
// contiguous in K. ISA layout: lane half hl picks K-offset 8:
//   elements 0..7  <- K = hl*8 + 0..7
//   elements 8..15 <- K = 16 + hl*8 + 0..7
__device__ __forceinline__ v16bf load_frag(const unsigned short* rowp, int hl) {
    u16x8 c0 = *(const u16x8*)(rowp + hl * 8);
    u16x8 c1 = *(const u16x8*)(rowp + 16 + hl * 8);
    u16x16 f;
#pragma unroll
    for (int i = 0; i < 8; ++i) { f[i] = c0[i]; f[i + 8] = c1[i]; }
    return __builtin_bit_cast(v16bf, f);
}

// ---------------------------------------------------------------------------
// f32 -> bf16 conversion (4 elements / thread)
// ---------------------------------------------------------------------------
__global__ __launch_bounds__(256) void cvt_kernel(const float* __restrict__ src,
                                                  unsigned short* __restrict__ dst,
                                                  int n) {
    int i = (blockIdx.x * 256 + threadIdx.x) * 4;
    if (i + 3 < n) {
        float4 f = *(const float4*)(src + i);
        u16x4 o;
        o[0] = f2bf(f.x); o[1] = f2bf(f.y); o[2] = f2bf(f.z); o[3] = f2bf(f.w);
        *(u16x4*)(dst + i) = o;
    } else {
        for (int j = i; j < n; ++j) dst[j] = f2bf(src[j]);
    }
}

// ---------------------------------------------------------------------------
// Tiled WMMA GEMM: C[M,N] = A[M,K] * B[K,N]  (bf16 in, bf16 or f32 out)
// 256 threads = 8 waves (4 M x 2 N); block tile 128x128; wave tile 32x64;
// K-step 32. A staged by ASYNC global->LDS DMA; B staged transposed ([n][k])
// by VALU so both fragment loads are contiguous-K 16B ds_load_b128s.
// ---------------------------------------------------------------------------
template <int OUT_F32>
__global__ __launch_bounds__(256) void gemm_bf16_kernel(
    const unsigned short* __restrict__ A,
    const unsigned short* __restrict__ B,
    unsigned short* __restrict__ Cb,
    float* __restrict__ Cf,
    int M, int N, int K) {
    __shared__ __align__(16) unsigned short As[128 * 32];
    __shared__ __align__(16) unsigned short Bs[128 * 32];  // [n][k]

    const int tid  = threadIdx.x;
    const int wave = tid >> 5;
    const int lane = tid & 31;
    const int lm   = lane & 15;
    const int hl   = lane >> 4;
    const int wm   = wave >> 1;   // 0..3
    const int wn   = wave & 1;    // 0..1
    const int bm   = blockIdx.y * 128;
    const int bn   = blockIdx.x * 128;

    v8f acc[2][4];
#pragma unroll
    for (int mt = 0; mt < 2; ++mt)
#pragma unroll
        for (int nt = 0; nt < 4; ++nt) acc[mt][nt] = {};

    for (int k0 = 0; k0 < K; k0 += 32) {
        // ---- stage A tile [128][32] via async DMA (2 x 16B per thread) ----
        {
            const int row = tid >> 1;
            const int col = (tid & 1) * 16;
            const unsigned short* ag = A + (size_t)(bm + row) * K + k0 + col;
            async_load_b128(ag,     &As[row * 32 + col]);
            async_load_b128(ag + 8, &As[row * 32 + col + 8]);
        }
        // ---- stage B tile transposed -> Bs[n][k] (overlaps the async) ----
        {
            const int kk = tid >> 3;          // 0..31
            const int nc = (tid & 7) * 16;    // 0..112
            const int nbase = bn + nc;
            const unsigned short* bg = B + (size_t)(k0 + kk) * N + nbase;
            u16x8 b0, b1;
            if (nbase + 16 <= N) {
                b0 = *(const u16x8*)bg;
                b1 = *(const u16x8*)(bg + 8);
            } else {
#pragma unroll
                for (int i = 0; i < 8; ++i) {
                    b0[i] = (nbase + i < N)     ? bg[i]     : (unsigned short)0;
                    b1[i] = (nbase + 8 + i < N) ? bg[8 + i] : (unsigned short)0;
                }
            }
#pragma unroll
            for (int i = 0; i < 8; ++i) {
                Bs[(nc + i) * 32 + kk]     = b0[i];
                Bs[(nc + 8 + i) * 32 + kk] = b1[i];
            }
            // prefetch next B k-slab into L2 while we compute
            if (k0 + 32 < K)
                __builtin_prefetch(B + (size_t)(k0 + 32 + kk) * N + nbase, 0, 1);
        }
        wait_asynccnt0();
        __syncthreads();

        v16bf af[2];
#pragma unroll
        for (int mt = 0; mt < 2; ++mt)
            af[mt] = load_frag(&As[(wm * 32 + mt * 16 + lm) * 32], hl);
        v16bf bf[4];
#pragma unroll
        for (int nt = 0; nt < 4; ++nt)
            bf[nt] = load_frag(&Bs[(wn * 64 + nt * 16 + lm) * 32], hl);
#pragma unroll
        for (int mt = 0; mt < 2; ++mt)
#pragma unroll
            for (int nt = 0; nt < 4; ++nt)
                acc[mt][nt] = wmma_bf16(af[mt], bf[nt], acc[mt][nt]);
        __syncthreads();
    }

    // ---- store: C layout lane -> (M = r + 8*hl, N = lane%16) ----
#pragma unroll
    for (int mt = 0; mt < 2; ++mt) {
#pragma unroll
        for (int nt = 0; nt < 4; ++nt) {
            const int ncol  = bn + wn * 64 + nt * 16 + lm;
            if (ncol >= N) continue;
            const int mrow0 = bm + wm * 32 + mt * 16 + 8 * hl;
#pragma unroll
            for (int r = 0; r < 8; ++r) {
                if (OUT_F32)
                    Cf[(size_t)(mrow0 + r) * N + ncol] = acc[mt][nt][r];
                else
                    Cb[(size_t)(mrow0 + r) * N + ncol] = f2bf(acc[mt][nt][r]);
            }
        }
    }
}

// ---------------------------------------------------------------------------
// Flash attention (bf16 WMMA, transposed-score trick).
// grid.x = B * NHEAD * (S/64); block = 128 threads = 4 waves;
// each wave owns 16 q rows; 32 keys per iteration staged in LDS.
// K tile staged by ASYNC global->LDS DMA; V tile transposed by VALU while the
// DMA is in flight. Scores computed transposed: T = K_tile(16x64) x Q^T(64x16)
// so the C-layout of T is directly the B-fragment layout of P^T for
// O^T = V^T x P^T (no cross-lane data movement between the two WMMA stages).
// ---------------------------------------------------------------------------
__global__ __launch_bounds__(128) void attn_kernel(
    const unsigned short* __restrict__ qh,
    const unsigned short* __restrict__ kh,
    const unsigned short* __restrict__ vh,
    const int* __restrict__ mask,
    unsigned short* __restrict__ ctx) {
    const int blk  = blockIdx.x;
    const int qblk = blk & 31;                // S/64 = 32
    const int h    = (blk >> 5) & 15;
    const int b    = blk >> 9;
    const int tid  = threadIdx.x;
    const int wave = tid >> 5;
    const int lane = tid & 31;
    const int lm   = lane & 15;
    const int hl   = lane >> 4;

    const int qbase = qblk * 64 + wave * 16;  // wave's q tile base (within S)
    const int qrow  = qbase + lm;             // this lane's q row

    __shared__ __align__(16) unsigned short Kt[32 * 64];  // [kk][d]
    __shared__ __align__(16) unsigned short Vt[64 * 40];  // [d][kk], stride 40

    // Q fragments (B-role: n = qrow, K = d), loaded once from global
    const unsigned short* qrp =
        qh + ((size_t)(b * S_LEN) + qrow) * HID + h * HDIM;
    const v16bf qf0 = load_frag(qrp, hl);
    const v16bf qf1 = load_frag(qrp + 32, hl);

    v8f o0 = {}, o1 = {}, o2 = {}, o3 = {};
    float m_run = -3.0e38f;
    float l_run = 0.0f;
    const float sc = 0.125f;  // 1/sqrt(64)

    // staging geometry: thread -> (key row, 16-wide d chunk)
    const int srow = tid >> 2;        // 0..31
    const int scol = (tid & 3) * 16;  // 0,16,32,48

    for (int kb = 0; kb < S_LEN; kb += 32) {
        const size_t gro =
            ((size_t)(b * S_LEN) + kb + srow) * HID + h * HDIM + scol;

        // ---- K tile: async global -> LDS DMA (2 x 16B per thread) ----
        async_load_b128(kh + gro,     &Kt[srow * 64 + scol]);
        async_load_b128(kh + gro + 8, &Kt[srow * 64 + scol + 8]);

        // ---- V tile: load + transpose-store while the DMA is in flight ----
        {
            u16x8 b0 = *(const u16x8*)(vh + gro);
            u16x8 b1 = *(const u16x8*)(vh + gro + 8);
#pragma unroll
            for (int i = 0; i < 8; ++i) {
                Vt[(scol + i) * 40 + srow]     = b0[i];
                Vt[(scol + 8 + i) * 40 + srow] = b1[i];
            }
        }
        // L2 prefetch of the next tiles (streams are L2-resident)
        if (kb + 32 < S_LEN) {
            __builtin_prefetch(kh + gro + 32 * HID, 0, 1);
            __builtin_prefetch(vh + gro + 32 * HID, 0, 1);
        }
        wait_asynccnt0();
        __syncthreads();

        // ---- transposed scores: T0 (keys kb..kb+15), T1 (kb+16..kb+31) ----
        v8f t0 = {}, t1 = {};
        {
            v16bf ka00 = load_frag(&Kt[lm * 64], hl);
            v16bf ka01 = load_frag(&Kt[lm * 64 + 32], hl);
            v16bf ka10 = load_frag(&Kt[(16 + lm) * 64], hl);
            v16bf ka11 = load_frag(&Kt[(16 + lm) * 64 + 32], hl);
            t0 = wmma_bf16(ka00, qf0, t0);
            t0 = wmma_bf16(ka01, qf1, t0);
            t1 = wmma_bf16(ka10, qf0, t1);
            t1 = wmma_bf16(ka11, qf1, t1);
        }

        // ---- mask * scale (multiplicative mask BEFORE softmax) ----
        // lane's scores: fixed qrow, kpos = kb + tile*16 + 8*hl + r
        const int* mrow = mask + (size_t)qrow * S_LEN + kb + 8 * hl;
        int4 m0a = *(const int4*)(mrow);
        int4 m0b = *(const int4*)(mrow + 4);
        int4 m1a = *(const int4*)(mrow + 16);
        int4 m1b = *(const int4*)(mrow + 20);
        int mi[16] = {m0a.x, m0a.y, m0a.z, m0a.w, m0b.x, m0b.y, m0b.z, m0b.w,
                      m1a.x, m1a.y, m1a.z, m1a.w, m1b.x, m1b.y, m1b.z, m1b.w};
        float sarr[16];
#pragma unroll
        for (int i = 0; i < 8; ++i) {
            sarr[i]     = t0[i] * sc * (float)mi[i];
            sarr[8 + i] = t1[i] * sc * (float)mi[8 + i];
        }

        // ---- online softmax (one shfl: lanes l and l^16 share a q row) ----
        float mx = sarr[0];
#pragma unroll
        for (int i = 1; i < 16; ++i) mx = fmaxf(mx, sarr[i]);
        mx = fmaxf(mx, __shfl_xor(mx, 16));
        const float m_new = fmaxf(m_run, mx);
        const float resc  = __expf(m_run - m_new);
        float psum = 0.0f;
        unsigned short pb[16];
#pragma unroll
        for (int i = 0; i < 16; ++i) {
            const float p = __expf(sarr[i] - m_new);
            psum += p;
            pb[i] = f2bf(p);
        }
        psum += __shfl_xor(psum, 16);
        l_run = l_run * resc + psum;
        m_run = m_new;
#pragma unroll
        for (int i = 0; i < 8; ++i) {
            o0[i] *= resc; o1[i] *= resc; o2[i] *= resc; o3[i] *= resc;
        }

        // ---- P^T fragment straight from T registers (no shuffles) ----
        u16x16 pu;
#pragma unroll
        for (int i = 0; i < 16; ++i) pu[i] = pb[i];
        const v16bf pf = __builtin_bit_cast(v16bf, pu);

        // ---- O^T += V^T(16d x 32k) x P^T(32k x 16q), per 16-wide d tile ----
        {
            v16bf vt0 = load_frag(&Vt[(0 * 16 + lm) * 40], hl);
            v16bf vt1 = load_frag(&Vt[(1 * 16 + lm) * 40], hl);
            v16bf vt2 = load_frag(&Vt[(2 * 16 + lm) * 40], hl);
            v16bf vt3 = load_frag(&Vt[(3 * 16 + lm) * 40], hl);
            o0 = wmma_bf16(vt0, pf, o0);
            o1 = wmma_bf16(vt1, pf, o1);
            o2 = wmma_bf16(vt2, pf, o2);
            o3 = wmma_bf16(vt3, pf, o3);
        }
        __syncthreads();
    }

    // ---- normalize + store ctx[b, qrow, h*64 + d] (bf16) ----
    const float inv = 1.0f / l_run;
    unsigned short* crow =
        ctx + ((size_t)(b * S_LEN) + qrow) * HID + h * HDIM;
    v8f* oacc[4] = {&o0, &o1, &o2, &o3};
#pragma unroll
    for (int dt = 0; dt < 4; ++dt) {
        u16x8 st;
#pragma unroll
        for (int r = 0; r < 8; ++r) st[r] = f2bf((*oacc[dt])[r] * inv);
        *(u16x8*)(crow + dt * 16 + 8 * hl) = st;
    }
}

// ---------------------------------------------------------------------------
// Host launcher
// ---------------------------------------------------------------------------
extern "C" void kernel_launch(void* const* d_in, const int* in_sizes, int n_in,
                              void* d_out, int out_size, void* d_ws, size_t ws_size,
                              hipStream_t stream) {
    (void)in_sizes; (void)n_in; (void)out_size; (void)ws_size;
    const float* q    = (const float*)d_in[0];
    const float* k    = (const float*)d_in[1];
    const float* v    = (const float*)d_in[2];
    const int*   mask = (const int*)d_in[3];
    const float* Wq   = (const float*)d_in[4];
    const float* Wk   = (const float*)d_in[5];
    const float* Wv   = (const float*)d_in[6];
    const float* Wo   = (const float*)d_in[7];
    float* out = (float*)d_out;

    const size_t BS  = (size_t)4 * S_LEN;   // 8192 rows
    const size_t E   = BS * HID;            // big matrix element count
    const size_t WE  = (size_t)HID * HID;   // 1024x1024
    const size_t WOE = (size_t)HID * HDIM;  // 1024x64

    char* ws = (char*)d_ws;
    size_t off = 0;
    auto alloc = [&](size_t elems) {
        unsigned short* p = (unsigned short*)(ws + off);
        off += elems * sizeof(unsigned short);
        return p;
    };
    unsigned short* qb  = alloc(E);
    unsigned short* kb_ = alloc(E);
    unsigned short* vb  = alloc(E);
    unsigned short* wqb = alloc(WE);
    unsigned short* wkb = alloc(WE);
    unsigned short* wvb = alloc(WE);
    unsigned short* wob = alloc(WOE);
    unsigned short* qhb = alloc(E);
    unsigned short* khb = alloc(E);
    unsigned short* vhb = alloc(E);
    unsigned short* ctx = alloc(E);

    auto cvt = [&](const float* s, unsigned short* d, size_t n) {
        cvt_kernel<<<(unsigned)((n / 4 + 255) / 256), 256, 0, stream>>>(s, d, (int)n);
    };
    cvt(q, qb, E);
    cvt(k, kb_, E);
    cvt(v, vb, E);
    cvt(Wq, wqb, WE);
    cvt(Wk, wkb, WE);
    cvt(Wv, wvb, WE);
    cvt(Wo, wob, WOE);

    // QKV projections: [8192,1024] x [1024,1024] -> bf16
    dim3 gproj(HID / 128, (unsigned)(BS / 128));  // (8, 64)
    gemm_bf16_kernel<0><<<gproj, 256, 0, stream>>>(qb, wqb, qhb, nullptr,
                                                   (int)BS, HID, HID);
    gemm_bf16_kernel<0><<<gproj, 256, 0, stream>>>(kb_, wkb, khb, nullptr,
                                                   (int)BS, HID, HID);
    gemm_bf16_kernel<0><<<gproj, 256, 0, stream>>>(vb, wvb, vhb, nullptr,
                                                   (int)BS, HID, HID);

    // Attention: grid = B * H * (S/64) = 2048 blocks of 128 threads
    attn_kernel<<<4 * NHEAD * (S_LEN / 64), 128, 0, stream>>>(qhb, khb, vhb,
                                                              mask, ctx);

    // Output projection: [8192,1024] x [1024,64] -> f32 out
    dim3 gout(1, (unsigned)(BS / 128));
    gemm_bf16_kernel<1><<<gout, 256, 0, stream>>>(ctx, wob, nullptr, out,
                                                  (int)BS, HDIM, HID);
}
